// NSAttention_12816182411653
// MI455X (gfx1250) — matrix operand order repaired and checked by reading, not compiled, not run on hardware
//
#include <hip/hip_runtime.h>
#include <hip/hip_bf16.h>

#define B_  2
#define S_  2048
#define H_  2048
#define NH_ 16
#define HD_ 128

typedef __bf16 bf16_t;
typedef __attribute__((ext_vector_type(16))) __bf16 bf16x16;
typedef __attribute__((ext_vector_type(8)))  __bf16 bf16x8;
typedef __attribute__((ext_vector_type(8)))  float  f32x8;
typedef __attribute__((ext_vector_type(4)))  unsigned int u32x4;
typedef __attribute__((ext_vector_type(4)))  int  i32x4;
typedef __attribute__((ext_vector_type(8)))  int  i32x8;

// ---------------------------------------------------------------------------
// WMMA helpers (CDNA5 wave32, V_WMMA_F32_16X16X32_BF16)
// ---------------------------------------------------------------------------
__device__ __forceinline__ f32x8 wmma_bf16(bf16x16 a, bf16x16 b, f32x8 c) {
  return __builtin_amdgcn_wmma_f32_16x16x32_bf16(false, a, false, b, (short)0, c,
                                                 false, false);
}

__device__ __forceinline__ f32x8 zero8() {
  f32x8 z;
#pragma unroll
  for (int i = 0; i < 8; ++i) z[i] = 0.0f;
  return z;
}

union FragU { bf16x16 v; bf16x8 h[2]; };

// A fragment 16x32 (MxK), src row-major, leading dim ld (elements).
__device__ __forceinline__ bf16x16 load_A16x32(const bf16_t* src, int ld) {
  const unsigned lane = threadIdx.x & 31u;
  const unsigned m    = lane & 15u;
  const unsigned kb   = (lane >> 4) * 8u;          // 0 or 8
  const bf16_t* p = src + (size_t)m * ld + kb;
  FragU u;
  u.h[0] = *(const bf16x8*)(p);                    // K = kb .. kb+7
  u.h[1] = *(const bf16x8*)(p + 16);               // K = kb+16 .. kb+23
  return u.v;
}

// B fragment 32x16 (KxN) where B[k][n] = src[n*ld + k] ("NT": src is N-major).
__device__ __forceinline__ bf16x16 load_B32x16(const bf16_t* src, int ld) {
  const unsigned lane = threadIdx.x & 31u;
  const unsigned n    = lane & 15u;
  const unsigned kb   = (lane >> 4) * 8u;          // 0 or 8
  const bf16_t* p = src + (size_t)n * ld + kb;
  FragU u;
  u.h[0] = *(const bf16x8*)(p);
  u.h[1] = *(const bf16x8*)(p + 16);
  return u.v;
}

__device__ __forceinline__ void store_out(float* p, float v)  { *p = v; }
__device__ __forceinline__ void store_out(bf16_t* p, float v) { *p = (bf16_t)v; }

// ---------------------------------------------------------------------------
// DPP-based 16-lane reductions (no LDS traffic):
//   xor1 = quad_perm(1,0,3,2); xor2 = quad_perm(2,3,0,1);
//   xor4 = DPP8 octet permute; xor8 = row_ror:8 (rot-8 in 16 lanes == xor 8)
// ---------------------------------------------------------------------------
__device__ __forceinline__ float dpp_xor1(float x) {
  return __int_as_float(
      __builtin_amdgcn_mov_dpp(__float_as_int(x), 0xB1, 0xF, 0xF, true));
}
__device__ __forceinline__ float dpp_xor2(float x) {
  return __int_as_float(
      __builtin_amdgcn_mov_dpp(__float_as_int(x), 0x4E, 0xF, 0xF, true));
}
__device__ __forceinline__ float dpp_xor4(float x) {
  constexpr int SEL = 4 | (5 << 3) | (6 << 6) | (7 << 9) |
                      (0 << 12) | (1 << 15) | (2 << 18) | (3 << 21);
  return __int_as_float(__builtin_amdgcn_mov_dpp8(__float_as_int(x), SEL));
}
__device__ __forceinline__ float dpp_xor8(float x) {
  return __int_as_float(
      __builtin_amdgcn_mov_dpp(__float_as_int(x), 0x128, 0xF, 0xF, true));
}
__device__ __forceinline__ float red16_max(float x) {
  x = fmaxf(x, dpp_xor1(x));
  x = fmaxf(x, dpp_xor2(x));
  x = fmaxf(x, dpp_xor4(x));
  x = fmaxf(x, dpp_xor8(x));
  return x;
}
__device__ __forceinline__ float red16_sum(float x) {
  x += dpp_xor1(x);
  x += dpp_xor2(x);
  x += dpp_xor4(x);
  x += dpp_xor8(x);
  return x;
}

// ---------------------------------------------------------------------------
// Tensor Data Mover: DMA a 128-row x 32-elem bf16 tile (row stride = strideE)
// from global to LDS. Hardware padding: 16 DWORDs (64B row) + 4 DWORDs pad
// -> 80B LDS row stride == LDSS elements.
// ---------------------------------------------------------------------------
#define LDSS 40

__device__ __forceinline__ void tdm_load_tile_128x32(unsigned ldsOff,
                                                     const bf16_t* gptr,
                                                     unsigned strideE) {
  unsigned long long ga = (unsigned long long)(size_t)gptr;
  u32x4 g0;
  g0[0] = 1u;                                        // count=1, user descriptor
  g0[1] = ldsOff;                                    // lds_addr (bytes)
  g0[2] = (unsigned)(ga & 0xFFFFFFFFu);              // global_addr[31:0]
  g0[3] = (unsigned)((ga >> 32) & 0x1FFFFFFu) | (2u << 30);  // addr[56:32]|type=2
  i32x8 g1;
  // data_size=2B (1<<16), pad_enable (1<<20), pad_interval=16DW (3<<22),
  // pad_amount=4DW (3<<25)
  g1[0] = (int)((1u << 16) | (1u << 20) | (3u << 22) | (3u << 25));
  g1[1] = (int)((strideE & 0xFFFFu) << 16);          // tensor_dim0[15:0]
  g1[2] = (int)(((strideE >> 16) & 0xFFFFu) |        // tensor_dim0[31:16]
                (128u << 16));                       // tensor_dim1[15:0] = 128
  g1[3] = (int)(32u << 16);                          // tile_dim0 = 32
  g1[4] = (int)128u;                                 // tile_dim1 = 128
  g1[5] = (int)strideE;                              // tensor_dim0_stride lo
  g1[6] = 0;
  g1[7] = 0;
  i32x4 z4;
  z4[0] = 0; z4[1] = 0; z4[2] = 0; z4[3] = 0;
  i32x8 z8;
#pragma unroll
  for (int i = 0; i < 8; ++i) z8[i] = 0;
  __builtin_amdgcn_tensor_load_to_lds(g0, g1, z4, z4, z8, 0);
}

// ---------------------------------------------------------------------------
// fp32 -> bf16 conversion
// ---------------------------------------------------------------------------
__global__ void f32_to_bf16(const float* __restrict__ in, bf16_t* __restrict__ out,
                            size_t n) {
  size_t i = (size_t)blockIdx.x * blockDim.x + threadIdx.x;
  if (i < n) out[i] = (bf16_t)in[i];
}

// ---------------------------------------------------------------------------
// Generic bf16 NT GEMM: C(MxN) = A(MxK) @ W(NxK)^T, fp32 accumulate.
// Block = 256 threads (8 waves), 128x128 C tile.
// TDM double-buffer pipeline: wave 0 issues the DMA for slab t+1, all waves
// compute slab t, wave 0 drains TENSORcnt just before the barrier.
// ---------------------------------------------------------------------------
template <typename OutT>
__global__ __launch_bounds__(256) void gemm_bf16_nt(const bf16_t* __restrict__ A,
                                                    const bf16_t* __restrict__ W,
                                                    OutT* __restrict__ C,
                                                    int M, int N, int K) {
  __shared__ __align__(16) bf16_t As[2][128 * LDSS];
  __shared__ __align__(16) bf16_t Ws[2][128 * LDSS];

  const int m0 = blockIdx.y * 128;
  const int n0 = blockIdx.x * 128;
  const unsigned tid  = threadIdx.x;
  const unsigned wave = tid >> 5;        // 0..7
  const unsigned wm   = wave >> 2;       // 0..1  -> 64 rows of C
  const unsigned wn   = wave & 3u;       // 0..3  -> 32 cols of C

  unsigned aOff[2], wOff[2];
  aOff[0] = (unsigned)(size_t)(void*)As[0];
  aOff[1] = (unsigned)(size_t)(void*)As[1];
  wOff[0] = (unsigned)(size_t)(void*)Ws[0];
  wOff[1] = (unsigned)(size_t)(void*)Ws[1];

  const bf16_t* Arow = A + (size_t)m0 * K;
  const bf16_t* Wrow = W + (size_t)n0 * K;

  f32x8 acc[4][2];
#pragma unroll
  for (int i = 0; i < 4; ++i)
#pragma unroll
    for (int j = 0; j < 2; ++j) acc[i][j] = zero8();

  // ---- prologue: fill buffer 0 ----
  if (wave == 0) {
    tdm_load_tile_128x32(aOff[0], Arow, (unsigned)K);
    tdm_load_tile_128x32(wOff[0], Wrow, (unsigned)K);
    __builtin_amdgcn_s_wait_tensorcnt(0);
  }
  __syncthreads();

  unsigned cur = 0;
  for (int kt = 0; kt < K; kt += 32) {
    // ---- issue async DMA for the next slab into the other buffer ----
    if (wave == 0 && kt + 32 < K) {
      tdm_load_tile_128x32(aOff[cur ^ 1u], Arow + kt + 32, (unsigned)K);
      tdm_load_tile_128x32(wOff[cur ^ 1u], Wrow + kt + 32, (unsigned)K);
    }

    // ---- compute current slab (DMA for next slab in flight) ----
    const bf16_t* as = As[cur];
    const bf16_t* wsb = Ws[cur];
    bf16x16 bfrag[2];
#pragma unroll
    for (int j = 0; j < 2; ++j)
      bfrag[j] = load_B32x16(wsb + (wn * 32 + j * 16) * LDSS, LDSS);
#pragma unroll
    for (int i = 0; i < 4; ++i) {
      bf16x16 afrag = load_A16x32(as + (wm * 64 + i * 16) * LDSS, LDSS);
#pragma unroll
      for (int j = 0; j < 2; ++j) acc[i][j] = wmma_bf16(afrag, bfrag[j], acc[i][j]);
    }

    // ---- drain DMA, then release everyone into the next slab ----
    if (wave == 0) __builtin_amdgcn_s_wait_tensorcnt(0);
    __syncthreads();
    cur ^= 1u;
  }

  // ---- epilogue: C layout = lane->column, VGPR r -> row r / r+8 ----
  const unsigned lane = tid & 31u;
  const unsigned cn   = lane & 15u;
  const unsigned rOff = (lane >> 4) * 8u;
#pragma unroll
  for (int i = 0; i < 4; ++i) {
#pragma unroll
    for (int j = 0; j < 2; ++j) {
#pragma unroll
      for (int r = 0; r < 8; ++r) {
        size_t row = (size_t)(m0 + wm * 64 + i * 16 + rOff + r);
        store_out(&C[row * N + (n0 + wn * 32 + j * 16 + cn)], acc[i][j][r]);
      }
    }
  }
}

// ---------------------------------------------------------------------------
// RoPE + layout: (B,S,H) -> Q,K (B,NH,S,HD) rotated; V transposed (B,NH,HD,S)
// ---------------------------------------------------------------------------
__global__ void rope_reshape(const bf16_t* __restrict__ Ql,
                             const bf16_t* __restrict__ Kl,
                             const bf16_t* __restrict__ Vl,
                             bf16_t* __restrict__ Q, bf16_t* __restrict__ Kr,
                             bf16_t* __restrict__ Vt) {
  size_t i = (size_t)blockIdx.x * blockDim.x + threadIdx.x;
  if (i >= (size_t)B_ * S_ * H_) return;
  int d  = (int)(i % HD_);
  int nh = (int)((i / HD_) % NH_);
  int s  = (int)((i / H_) % S_);
  int b  = (int)(i / ((size_t)S_ * H_));

  float q = (float)Ql[i], k = (float)Kl[i], v = (float)Vl[i];

  int   j    = d & 63;
  float invf = __powf(10000.0f, -(float)j / 64.0f);
  float sn, cs;
  __sincosf((float)s * invf, &sn, &cs);

  int    d2  = (d < 64) ? d + 64 : d - 64;
  float  sgn = (d < 64) ? -1.0f : 1.0f;
  size_t i2  = i - (size_t)d + (size_t)d2;
  float  qr  = sgn * (float)Ql[i2];
  float  kr  = sgn * (float)Kl[i2];

  size_t oq = (((size_t)b * NH_ + nh) * S_ + s) * HD_ + d;
  Q[oq]  = (bf16_t)(q * cs + qr * sn);
  Kr[oq] = (bf16_t)(k * cs + kr * sn);
  Vt[(((size_t)b * NH_ + nh) * HD_ + d) * S_ + s] = (bf16_t)v;
}

// ---------------------------------------------------------------------------
// Flash attention: per wave 16 query rows, 32-key blocks, online softmax.
// Q,K: (B,NH,S,HD)  V: transposed (B,NH,HD,S)  out: (B,S,H) bf16
// ---------------------------------------------------------------------------
__global__ __launch_bounds__(256) void attn_fa(const bf16_t* __restrict__ Q,
                                               const bf16_t* __restrict__ Kd,
                                               const bf16_t* __restrict__ Vt,
                                               bf16_t* __restrict__ O) {
  const int h  = blockIdx.y;
  const int bz = blockIdx.z;
  const int m0 = blockIdx.x * 128;
  const unsigned tid  = threadIdx.x;
  const unsigned wave = tid >> 5;
  const unsigned lane = tid & 31u;
  const unsigned cn   = lane & 15u;
  const unsigned rOff = (lane >> 4) * 8u;

  const bf16_t* Qh = Q  + ((size_t)(bz * NH_ + h)) * S_ * HD_;
  const bf16_t* Kh = Kd + ((size_t)(bz * NH_ + h)) * S_ * HD_;
  const bf16_t* Vh = Vt + ((size_t)(bz * NH_ + h)) * HD_ * S_;

  const int mw = m0 + (int)wave * 16;

  __shared__ __align__(16) bf16_t Pb[8][16 * LDSS];   // per-wave 16x32 P tile

  bf16x16 qf[4];
#pragma unroll
  for (int d = 0; d < 4; ++d)
    qf[d] = load_A16x32(Qh + (size_t)mw * HD_ + d * 32, HD_);

  f32x8 o[8];
#pragma unroll
  for (int j = 0; j < 8; ++j) o[j] = zero8();
  float m_r[8], l_r[8];
#pragma unroll
  for (int r = 0; r < 8; ++r) { m_r[r] = -1e30f; l_r[r] = 0.0f; }

  const float scale = 0.08838834764831845f;  // 1/sqrt(128)

  for (int kb = 0; kb < S_; kb += 32) {
    // ---- S = Q @ K^T for 32 keys (two 16x16 tiles, K-dim = 128) ----
    f32x8 s0 = zero8(), s1 = zero8();
#pragma unroll
    for (int d = 0; d < 4; ++d) {
      bf16x16 kf0 = load_B32x16(Kh + (size_t)kb * HD_ + d * 32, HD_);
      bf16x16 kf1 = load_B32x16(Kh + (size_t)(kb + 16) * HD_ + d * 32, HD_);
      s0 = wmma_bf16(qf[d], kf0, s0);
      s1 = wmma_bf16(qf[d], kf1, s1);
    }
    if (kb + 32 < S_) {
      __builtin_prefetch(Kh + (size_t)(kb + 32) * HD_ + (lane << 3), 0, 1);
      __builtin_prefetch(Vh + (size_t)cn * S_ + kb + 32, 0, 1);
    }

    // ---- online softmax (DPP reductions inside 16-lane groups) ----
#pragma unroll
    for (int r = 0; r < 8; ++r) {
      float a = s0[r] * scale;
      float b = s1[r] * scale;
      float mx    = red16_max(fmaxf(a, b));
      float mnew  = fmaxf(m_r[r], mx);
      float alpha = __expf(m_r[r] - mnew);
      float p0 = __expf(a - mnew);
      float p1 = __expf(b - mnew);
      float rs = red16_sum(p0 + p1);
      l_r[r] = l_r[r] * alpha + rs;
      m_r[r] = mnew;
#pragma unroll
      for (int j = 0; j < 8; ++j) o[j][r] *= alpha;
      // stash P in bf16 so it can be reloaded in A-fragment layout
      Pb[wave][(rOff + r) * LDSS + cn]      = (bf16_t)p0;
      Pb[wave][(rOff + r) * LDSS + 16 + cn] = (bf16_t)p1;
    }
    __syncthreads();

    // ---- O += P(16x32) @ V(32x128): 8 WMMAs over HD ----
    bf16x16 pf = load_A16x32(&Pb[wave][0], LDSS);
#pragma unroll
    for (int j = 0; j < 8; ++j) {
      bf16x16 vf = load_B32x16(Vh + (size_t)(j * 16) * S_ + kb, S_);
      o[j] = wmma_bf16(pf, vf, o[j]);
    }
    __syncthreads();
  }

  // ---- normalize and write (B,S,H) bf16 ----
#pragma unroll
  for (int r = 0; r < 8; ++r) {
    float inv = 1.0f / l_r[r];
    int s = mw + (int)rOff + r;
#pragma unroll
    for (int j = 0; j < 8; ++j)
      O[((size_t)bz * S_ + s) * H_ + h * HD_ + j * 16 + cn] =
          (bf16_t)(o[j][r] * inv);
  }
}

// ---------------------------------------------------------------------------
// Host launch
// ---------------------------------------------------------------------------
extern "C" void kernel_launch(void* const* d_in, const int* in_sizes, int n_in,
                              void* d_out, int out_size, void* d_ws, size_t ws_size,
                              hipStream_t stream) {
  const float* hs = (const float*)d_in[0];
  const float* Wq = (const float*)d_in[1];
  const float* Wk = (const float*)d_in[2];
  const float* Wv = (const float*)d_in[3];
  const float* Wo = (const float*)d_in[4];

  const size_t XE = (size_t)B_ * S_ * H_;   // 8,388,608 elems
  const size_t WE = (size_t)H_ * H_;        // 4,194,304 elems

  bf16_t* ws   = (bf16_t*)d_ws;
  bf16_t* Xb   = ws;
  bf16_t* Wqb  = Xb + XE;
  bf16_t* Wkb  = Wqb + WE;
  bf16_t* Wvb  = Wkb + WE;
  bf16_t* Wob  = Wvb + WE;
  bf16_t* Qlin = Wob + WE;      // reused as attention output buffer later
  bf16_t* Klin = Qlin + XE;
  bf16_t* Vlin = Klin + XE;
  bf16_t* Qr   = Vlin + XE;
  bf16_t* Kr   = Qr + XE;
  bf16_t* Vt   = Kr + XE;

  f32_to_bf16<<<(unsigned)((XE + 255) / 256), 256, 0, stream>>>(hs, Xb, XE);
  f32_to_bf16<<<(unsigned)((WE + 255) / 256), 256, 0, stream>>>(Wq, Wqb, WE);
  f32_to_bf16<<<(unsigned)((WE + 255) / 256), 256, 0, stream>>>(Wk, Wkb, WE);
  f32_to_bf16<<<(unsigned)((WE + 255) / 256), 256, 0, stream>>>(Wv, Wvb, WE);
  f32_to_bf16<<<(unsigned)((WE + 255) / 256), 256, 0, stream>>>(Wo, Wob, WE);

  dim3 gg(H_ / 128, (B_ * S_) / 128);   // (16, 32)
  gemm_bf16_nt<bf16_t><<<gg, 256, 0, stream>>>(Xb, Wqb, Qlin, B_ * S_, H_, H_);
  gemm_bf16_nt<bf16_t><<<gg, 256, 0, stream>>>(Xb, Wkb, Klin, B_ * S_, H_, H_);
  gemm_bf16_nt<bf16_t><<<gg, 256, 0, stream>>>(Xb, Wvb, Vlin, B_ * S_, H_, H_);

  rope_reshape<<<(unsigned)((XE + 255) / 256), 256, 0, stream>>>(Qlin, Klin, Vlin,
                                                                 Qr, Kr, Vt);

  dim3 ga(S_ / 128, NH_, B_);
  attn_fa<<<ga, 256, 0, stream>>>(Qr, Kr, Vt, Qlin);

  gemm_bf16_nt<float><<<gg, 256, 0, stream>>>(Qlin, Wob, (float*)d_out,
                                              B_ * S_, H_, H_);
}
